// SSIMLoss_11544872092177
// MI455X (gfx1250) — matrix operand order repaired and checked by reading, not compiled, not run on hardware
//
#include <hip/hip_runtime.h>

typedef __attribute__((ext_vector_type(16))) _Float16     v16h;
typedef __attribute__((ext_vector_type(8)))  float        v8f;
typedef __attribute__((ext_vector_type(4)))  unsigned int u32x4;
typedef __attribute__((ext_vector_type(8)))  int          i32x8;
typedef __attribute__((ext_vector_type(4)))  int          i32x4;

#define TILE 16          // output tile (H x W)
#define INW  22          // input tile side = TILE + 6
#define ROWS 32          // padded row count of LDS plane tile (rows 22..31 == 0)
#define TDIM 32
#define HDIM 320
#define WDIM 320
#define HOUT 314
#define WOUT 314
#define TOUT 26

#if __has_builtin(__builtin_amdgcn_tensor_load_to_lds)
#define HAVE_TDM 1
#endif

#ifdef HAVE_TDM
// Issue one TDM 2-D tile load: 22x22 f32 tile at (h0,w0) of a 320x320 plane,
// remaining-extent dims so OOB elements are zero-filled (edge padding for free).
__device__ __forceinline__ void tdm_issue_2d(const float* gptr, unsigned lds_off,
                                             int rem_w, int rem_h)
{
    unsigned long long ga = (unsigned long long)gptr;
    u32x4 g0;
    g0[0] = 1u;                                           // count=1, user descriptor
    g0[1] = lds_off;                                      // LDS dest (bytes)
    g0[2] = (unsigned)(ga & 0xffffffffu);                 // global_addr[31:0]
    g0[3] = (unsigned)((ga >> 32) & 0x01ffffffu)          // global_addr[56:32]
          | (2u << 30);                                   // type = 2 ("image")
    i32x8 g1;
    g1[0] = (2 << 16);                                    // data_size = 4 bytes
    g1[1] = (int)((unsigned)rem_w << 16);                 // tensor_dim0[15:0]
    g1[2] = (int)(((unsigned)rem_h & 0xffffu) << 16);     // dim0[31:16]=0 | dim1[15:0]
    g1[3] = (int)((INW << 16) | ((unsigned)rem_h >> 16)); // dim1[31:16]=0 | tile_dim0=22
    g1[4] = INW;                                          // tile_dim1=22, tile_dim2=0
    g1[5] = WDIM;                                         // tensor_dim0_stride[31:0]
    g1[6] = (int)(((unsigned)(WDIM * HDIM) & 0xffffu) << 16); // stride0[47:32]=0|stride1[15:0]
    g1[7] = (int)((unsigned)(WDIM * HDIM) >> 16);         // stride1[47:16]
    i32x4 gz = {0, 0, 0, 0};
#if __clang_major__ >= 23
    i32x8 gz8 = {0, 0, 0, 0, 0, 0, 0, 0};
    __builtin_amdgcn_tensor_load_to_lds(g0, g1, gz, gz, gz8, 0);
#else
    __builtin_amdgcn_tensor_load_to_lds(g0, g1, gz, gz, 0);
#endif
}
#endif

// Issue (or perform) the staging of both input plane tiles for plane t.
__device__ __forceinline__ void issue_plane(const float* __restrict__ pred,
                                            const float* __restrict__ targ,
                                            float* dstx, float* dsty,
                                            int b, int t, int h0, int w0)
{
    size_t base = (((size_t)b * TDIM + t) * HDIM + h0) * WDIM + w0;
#ifdef HAVE_TDM
    tdm_issue_2d(pred + base, (unsigned)(unsigned long long)(void*)dstx,
                 WDIM - w0, HDIM - h0);
    tdm_issue_2d(targ + base, (unsigned)(unsigned long long)(void*)dsty,
                 WDIM - w0, HDIM - h0);
#else
    for (int i = threadIdx.x; i < INW * INW; i += 32) {
        int r = i / INW, c = i - r * INW;
        int hg = h0 + r, wg = w0 + c;
        float xv = 0.0f, yv = 0.0f;
        if (hg < HDIM && wg < WDIM) {
            xv = pred[base + (size_t)r * WDIM + c];
            yv = targ[base + (size_t)r * WDIM + c];
        }
        dstx[i] = xv;
        dsty[i] = yv;
    }
#endif
}

#ifdef HAVE_TDM
#define WAIT_PLANES(n) __builtin_amdgcn_s_wait_tensorcnt(n)
#else
#define WAIT_PLANES(n) __syncthreads()
#endif

// One wave (32 threads) per block; streams all 32 T-planes of one
// (batch, 16x16 HxW) tile. Per plane: five 2-D 7x7 box sums via 3 WMMAs each
// (band-matrix matmuls on the matrix pipe), T-axis 7-window via LDS ring.
// Plane tiles are TDM double-buffered (fetch t+1 while computing t).
__global__ __launch_bounds__(32) void ssim3d_fused_kernel(
    const float* __restrict__ pred, const float* __restrict__ targ,
    const float* __restrict__ drange, float* __restrict__ partial)
{
    const int lane = threadIdx.x;        // 0..31
    const int half = lane >> 4;          // 0 or 1
    const int mrow = lane & 15;          // 0..15
    const int w0 = blockIdx.x * TILE;
    const int h0 = blockIdx.y * TILE;
    const int b  = blockIdx.z;

    __shared__ float    sxb[2][ROWS * INW];   // pred plane tiles (double buffer)
    __shared__ float    syb[2][ROWS * INW];   // targ plane tiles
    __shared__ _Float16 qbuf[16 * 32];        // Q (16x22, stride 32, zero pad)
    __shared__ float    ring[7][5][256];      // 7-plane ring of 2-D box sums
    __shared__ float    red[32];

    // zero qbuf (K columns 22..31 of pass-2 A) and padded rows 22..31 of tiles
    for (int i = lane; i < 16 * 32; i += 32) qbuf[i] = (_Float16)0.0f;
    for (int i = INW * INW + lane; i < ROWS * INW; i += 32) {
        sxb[0][i] = 0.0f; sxb[1][i] = 0.0f;
        syb[0][i] = 0.0f; syb[1][i] = 0.0f;
    }
    __syncthreads();

    // Pass-1 A fragment (constant): A[m][k] = 1 iff m <= k <= m+6   (16x32 f16 A layout)
    v16h bandA;
    #pragma unroll
    for (int j = 0; j < 8; ++j) {
        int kb = ((j < 4) ? 0 : 16) + 8 * half + 2 * (j & 3);
        bandA[2 * j]     = (_Float16)(((kb    ) >= mrow && (kb    ) <= mrow + 6) ? 1.0f : 0.0f);
        bandA[2 * j + 1] = (_Float16)(((kb + 1) >= mrow && (kb + 1) <= mrow + 6) ? 1.0f : 0.0f);
    }
    // Pass-2 B fragment (constant): B[k][n] = 1 iff n <= k <= n+6   (32x16 f16 B layout)
    v16h bandB;
    #pragma unroll
    for (int j = 0; j < 8; ++j) {
        int k0 = ((j < 4) ? 0 : 16) + 8 * half + 2 * (j & 3);
        bandB[2 * j]     = (_Float16)(((k0    ) >= mrow && (k0    ) <= mrow + 6) ? 1.0f : 0.0f);
        bandB[2 * j + 1] = (_Float16)(((k0 + 1) >= mrow && (k0 + 1) <= mrow + 6) ? 1.0f : 0.0f);
    }

    float acc[5][8];
    #pragma unroll
    for (int q = 0; q < 5; ++q)
        #pragma unroll
        for (int j = 0; j < 8; ++j) acc[q][j] = 0.0f;

    const float dr = drange[b];
    const float C1 = (0.01f * dr) * (0.01f * dr);
    const float C2 = (0.03f * dr) * (0.03f * dr);
    float loss = 0.0f;

    // prime the pipeline: plane 0 into buffer 0
    issue_plane(pred, targ, sxb[0], syb[0], b, 0, h0, w0);

    for (int t = 0; t < TDIM; ++t) {
        const int bb = t & 1;
        if (t + 1 < TDIM) {
            issue_plane(pred, targ, sxb[bb ^ 1], syb[bb ^ 1], b, t + 1, h0, w0);
            WAIT_PLANES(2);   // the 2 newest (t+1) may still be in flight
        } else {
            WAIT_PLANES(0);
        }
        __syncthreads();

        const float* sx = sxb[bb];
        const float* sy = syb[bb];
        const int slot = t % 7;
        v8f zero8 = {};

        #pragma unroll
        for (int q = 0; q < 5; ++q) {
            // Pass-1 B fragments: columns w = mrow (frag a) and w = mrow+6 (frag b)
            // of the product plane P[k][w], K = H rows (rows 22..31 are zero-padded).
            v16h fba, fbb;
            #pragma unroll
            for (int kk = 0; kk < 16; ++kk) {
                int k = (kk < 8) ? (8 * half + kk) : (16 + 8 * half + (kk - 8));
                float xa = sx[k * INW + mrow],     ya = sy[k * INW + mrow];
                float xb = sx[k * INW + mrow + 6], yb = sy[k * INW + mrow + 6];
                float pa, pb;
                if      (q == 0) { pa = xa;      pb = xb;      }
                else if (q == 1) { pa = ya;      pb = yb;      }
                else if (q == 2) { pa = xa * xa; pb = xb * xb; }
                else if (q == 3) { pa = ya * ya; pb = yb * yb; }
                else             { pa = xa * ya; pb = xb * yb; }
                fba[kk] = (_Float16)pa;
                fbb[kk] = (_Float16)pb;
            }
            // Pass 1: Q = Band^T x P  (7-tap sum along H), two column groups
            v8f d1a = __builtin_amdgcn_wmma_f32_16x16x32_f16(false, bandA, false, fba,
                                                             (short)0, zero8, false, false);
            v8f d1b = __builtin_amdgcn_wmma_f32_16x16x32_f16(false, bandA, false, fbb,
                                                             (short)0, zero8, false, false);
            // Spill Q (16x22 f16, row stride 32) to LDS for D -> A relayout
            #pragma unroll
            for (int j = 0; j < 8; ++j) {
                int m = j + 8 * half;
                qbuf[m * 32 + mrow] = (_Float16)d1a[j];
            }
            if (mrow >= 10) {
                #pragma unroll
                for (int j = 0; j < 8; ++j) {
                    int m = j + 8 * half;
                    qbuf[m * 32 + 6 + mrow] = (_Float16)d1b[j];
                }
            }
            __syncthreads();
            // Reload Q as A fragment
            v16h fa;
            #pragma unroll
            for (int j = 0; j < 8; ++j) {
                int kb = ((j < 4) ? 0 : 16) + 8 * half + 2 * (j & 3);
                fa[2 * j]     = qbuf[mrow * 32 + kb];
                fa[2 * j + 1] = qbuf[mrow * 32 + kb + 1];
            }
            // Pass 2: S = Q x Band (7-tap sum along W) -> 2-D 7x7 box sums
            v8f d2 = __builtin_amdgcn_wmma_f32_16x16x32_f16(false, fa, false, bandB,
                                                            (short)0, zero8, false, false);
            __syncthreads();

            // Sliding T-window: subtract plane t-7, add plane t (lane-private slots)
            float* rp = &ring[slot][q][lane * 8];
            #pragma unroll
            for (int j = 0; j < 8; ++j) {
                float nv = d2[j];
                float ov = (t >= 7) ? rp[j] : 0.0f;
                acc[q][j] += nv - ov;
                rp[j] = nv;
            }
        }

        if (t >= 6) {
            // acc[*] holds full 7x7x7 sums for t_out = t-6
            const float inv = 1.0f / 343.0f;
            const float cov = 343.0f / 342.0f;
            #pragma unroll
            for (int j = 0; j < 8; ++j) {
                int m = j + 8 * half;
                if ((h0 + m) < HOUT && (w0 + mrow) < WOUT) {
                    float ux  = acc[0][j] * inv, uy  = acc[1][j] * inv;
                    float uxx = acc[2][j] * inv, uyy = acc[3][j] * inv, uxy = acc[4][j] * inv;
                    float vx  = cov * (uxx - ux * ux);
                    float vy  = cov * (uyy - uy * uy);
                    float vxy = cov * (uxy - ux * uy);
                    float A1 = 2.0f * ux * uy + C1, A2 = 2.0f * vxy + C2;
                    float B1 = ux * ux + uy * uy + C1, B2 = vx + vy + C2;
                    loss += 1.0f - (A1 * A2) / (B1 * B2);
                }
            }
        }
        __syncthreads();
    }

    // wave-level partial reduction, one value per block (deterministic order)
    red[lane] = loss;
    __syncthreads();
    if (lane == 0) {
        float s = 0.0f;
        for (int i = 0; i < 32; ++i) s += red[i];
        partial[(blockIdx.z * gridDim.y + blockIdx.y) * gridDim.x + blockIdx.x] = s;
    }
}

__global__ void ssim3d_reduce_kernel(const float* __restrict__ partial,
                                     float* __restrict__ out, int n, float inv_count)
{
    __shared__ float s[256];
    float v = 0.0f;
    for (int i = threadIdx.x; i < n; i += 256) v += partial[i];
    s[threadIdx.x] = v;
    __syncthreads();
    for (int off = 128; off > 0; off >>= 1) {
        if ((int)threadIdx.x < off) s[threadIdx.x] += s[threadIdx.x + off];
        __syncthreads();
    }
    if (threadIdx.x == 0) out[0] = s[0] * inv_count;
}

extern "C" void kernel_launch(void* const* d_in, const int* in_sizes, int n_in,
                              void* d_out, int out_size, void* d_ws, size_t ws_size,
                              hipStream_t stream) {
    const float* pred = (const float*)d_in[0];
    const float* targ = (const float*)d_in[1];
    const float* dr   = (const float*)d_in[2];
    float* out        = (float*)d_out;
    float* partial    = (float*)d_ws;

    dim3 grid((WOUT + TILE - 1) / TILE,   // 20
              (HOUT + TILE - 1) / TILE,   // 20
              4);                         // batch
    ssim3d_fused_kernel<<<grid, 32, 0, stream>>>(pred, targ, dr, partial);

    const int nblk = grid.x * grid.y * grid.z;  // 1600
    const float inv_count = 1.0f / (4.0f * (float)TOUT * (float)HOUT * (float)WOUT);
    ssim3d_reduce_kernel<<<1, 256, 0, stream>>>(partial, out, nblk, inv_count);
}
